// ReversibleAttentionLayer_15066745274477
// MI455X (gfx1250) — matrix operand order repaired and checked by reading, not compiled, not run on hardware
//
#include <hip/hip_runtime.h>
#include <hip/hip_bf16.h>

typedef __attribute__((ext_vector_type(16))) __bf16 v16bf;
typedef __attribute__((ext_vector_type(8)))  float  v8f;
typedef unsigned int v4u  __attribute__((ext_vector_type(4)));
typedef int          v8i_ __attribute__((ext_vector_type(8)));
typedef int          v4i_ __attribute__((ext_vector_type(4)));

#define L_SEQ   2048
#define HALFD   1024
#define NHEADS  16
#define HDIM    64
#define MTOT    4096              // B*L
#define HBELEMS 4194304ull        // elems per Q/K/V/O buffer (32*2048*64)

#if defined(__HIP_DEVICE_COMPILE__) && __has_builtin(__builtin_amdgcn_tensor_load_to_lds)
#define HAVE_TDM 1
#else
#define HAVE_TDM 0
#endif

// ---------------- WMMA helpers (CDNA5 16x16x32 bf16, f32 accum) ----------------
static __device__ inline v8f wmma_bf16(v16bf a, v16bf b, v8f c) {
  return __builtin_amdgcn_wmma_f32_16x16x32_bf16(false, a, false, b, (short)0, c,
                                                 false, false);
}

// A fragment 16x32: lane = kHalf*16 + m ; elem e -> K = kHalf*16+e
static __device__ inline v16bf load_a_frag(const __bf16* __restrict__ base, int ld) {
  int lane = threadIdx.x & 31;
  int m = lane & 15, kh = lane >> 4;
  const __bf16* p = base + (size_t)m * ld;
  v16bf a;
#pragma unroll
  for (int e = 0; e < 8; ++e) a[e] = p[kh * 16 + e];
#pragma unroll
  for (int e = 0; e < 8; ++e) a[8 + e] = p[kh * 16 + 8 + e];
  return a;
}

// B fragment 32x16 stored LDS as [n][k]: lane = kHalf*16 + n ; elem e -> K = kHalf*16+e
static __device__ inline v16bf load_b_frag(const __bf16* __restrict__ base, int ld) {
  int lane = threadIdx.x & 31;
  int n = lane & 15, kh = lane >> 4;
  const __bf16* p = base + (size_t)n * ld + kh * 16;
  v16bf b;
#pragma unroll
  for (int e = 0; e < 16; ++e) b[e] = p[e];
  return b;
}

static __device__ inline v8f v8f_zero() {
  v8f z = {0.f, 0.f, 0.f, 0.f, 0.f, 0.f, 0.f, 0.f};
  return z;
}

// ================= Kernel 1: fused QKV projections (6 GEMMs) ====================
// Y = X(4096x1024) @ W^T(1024x1024). Block tile 128x128, BK=32, 8 waves, wave
// tile 32x64 (8 WMMA / wave / K-step). Register-pipelined global->LDS staging.
// grid = (1024/128, 4096/128, 6), block = 256.
#define BM 128
#define BN 128
#define BK 32
#define LDS_STRIDE 40

__global__ __launch_bounds__(256) void qkv_proj_kernel(
    const float* __restrict__ xa, const float* __restrict__ xb,
    const float* __restrict__ Wqa, const float* __restrict__ Wka,
    const float* __restrict__ Wva, const float* __restrict__ Wqb,
    const float* __restrict__ Wkb, const float* __restrict__ Wvb,
    __bf16* __restrict__ qkv) {
  __shared__ __align__(16) __bf16 As[BM][LDS_STRIDE];
  __shared__ __align__(16) __bf16 Bs[BN][LDS_STRIDE];

  const int z = blockIdx.z;  // 0..5: q_a,k_a,v_a,q_b,k_b,v_b
  const int branch = z / 3, which = z % 3;
  const float* X = branch ? xb : xa;
  const float* W;
  if (branch == 0) W = (which == 0) ? Wqa : ((which == 1) ? Wka : Wva);
  else             W = (which == 0) ? Wqb : ((which == 1) ? Wkb : Wvb);
  __bf16* out = qkv + (size_t)z * HBELEMS;

  const int m0 = blockIdx.y * BM;
  const int n0 = blockIdx.x * BN;
  const int tid = threadIdx.x, lane = tid & 31, wave = tid >> 5;
  const int wm = wave >> 1, wn = wave & 1;  // 4x2 grid of 32x64 wave tiles

  v8f acc[2][4];
#pragma unroll
  for (int i = 0; i < 2; ++i)
#pragma unroll
    for (int j = 0; j < 4; ++j) acc[i][j] = v8f_zero();

  const int tcol = tid & 7;   // 8 float4 per 32-float row
  const int trow = tid >> 3;  // 32 rows per pass

  float4 ra[4], rb[4];
  auto load_stage = [&](int kc) {
#pragma unroll
    for (int p = 0; p < 4; ++p) {
      ra[p] = *(const float4*)(X + (size_t)(m0 + trow + p * 32) * HALFD + kc + tcol * 4);
      rb[p] = *(const float4*)(W + (size_t)(n0 + trow + p * 32) * HALFD + kc + tcol * 4);
    }
  };
  auto store_stage = [&]() {
#pragma unroll
    for (int p = 0; p < 4; ++p) {
      __bf16* da = &As[trow + p * 32][tcol * 4];
      da[0] = (__bf16)ra[p].x; da[1] = (__bf16)ra[p].y;
      da[2] = (__bf16)ra[p].z; da[3] = (__bf16)ra[p].w;
      __bf16* db = &Bs[trow + p * 32][tcol * 4];
      db[0] = (__bf16)rb[p].x; db[1] = (__bf16)rb[p].y;
      db[2] = (__bf16)rb[p].z; db[3] = (__bf16)rb[p].w;
    }
  };

  load_stage(0);
  store_stage();
  __syncthreads();

  for (int k0 = 0; k0 < HALFD; k0 += BK) {
    const bool more = (k0 + BK) < HALFD;
    if (more) load_stage(k0 + BK);  // overlap next tile's loads with WMMAs

    v16bf bfr[4];
#pragma unroll
    for (int ni = 0; ni < 4; ++ni)
      bfr[ni] = load_b_frag(&Bs[wn * 64 + ni * 16][0], LDS_STRIDE);
#pragma unroll
    for (int mi = 0; mi < 2; ++mi) {
      v16bf a = load_a_frag(&As[wm * 32 + mi * 16][0], LDS_STRIDE);
#pragma unroll
      for (int ni = 0; ni < 4; ++ni)
        acc[mi][ni] = wmma_bf16(a, bfr[ni], acc[mi][ni]);
    }
    __syncthreads();
    if (more) {
      store_stage();
      __syncthreads();
    }
  }

  // epilogue: C layout n=lane&15, m=r+8*hi ; scatter to head-major bf16
  const int nl = lane & 15, hi = lane >> 4;
#pragma unroll
  for (int mi = 0; mi < 2; ++mi)
#pragma unroll
    for (int ni = 0; ni < 4; ++ni)
#pragma unroll
      for (int r = 0; r < 8; ++r) {
        int gm = m0 + wm * 32 + mi * 16 + r + 8 * hi;
        int gn = n0 + wn * 64 + ni * 16 + nl;
        int bb = gm >> 11, l = gm & 2047, h = gn >> 6, d = gn & 63;
        out[(((size_t)bb * NHEADS + h) * L_SEQ + l) * HDIM + d] =
            (__bf16)acc[mi][ni][r];
      }
}

// ================= Kernel 2: streaming cross-attention ==========================
// grid = (L/128, B*H, 2 branches), block = 256 (8 waves, 16 q-rows each).
// V tile staged to LDS once per key-block, shared by all 8 waves.
__global__ __launch_bounds__(256) void cross_attn_kernel(
    const __bf16* __restrict__ qkv, __bf16* __restrict__ obuf) {
  const int branch = blockIdx.z;
  const int bh = blockIdx.y;
  const int qb = blockIdx.x * 128;
  const __bf16* Q = qkv + (branch ? 3 * HBELEMS : 0) + (size_t)bh * L_SEQ * HDIM;
  const __bf16* K = qkv + (branch ? 1 * HBELEMS : 4 * HBELEMS) + (size_t)bh * L_SEQ * HDIM;
  const __bf16* V = qkv + (branch ? 2 * HBELEMS : 5 * HBELEMS) + (size_t)bh * L_SEQ * HDIM;

  const int tid = threadIdx.x;
  const int lane = tid & 31, wave = tid >> 5;
  const int qrow0 = qb + wave * 16;
  const int nl = lane & 15, hi = lane >> 4;
  const float inv_scale = 0.08838834764831845f;  // 1/sqrt(128)

  __shared__ __align__(16) __bf16 plds[8][16][40];
  __shared__ __align__(16) __bf16 Vs[32][72];

  // Q fragments, resident all kernel (d split 0..31 / 32..63)
  v16bf qf[2];
#pragma unroll
  for (int ds = 0; ds < 2; ++ds) {
    const __bf16* p = Q + (size_t)(qrow0 + nl) * HDIM + ds * 32;
#pragma unroll
    for (int e = 0; e < 8; ++e) qf[ds][e] = p[hi * 16 + e];
#pragma unroll
    for (int e = 0; e < 8; ++e) qf[ds][8 + e] = p[hi * 16 + 8 + e];
  }

  v8f o[4];
#pragma unroll
  for (int i = 0; i < 4; ++i) o[i] = v8f_zero();
  float mrun[8], lrun[8];
#pragma unroll
  for (int r = 0; r < 8; ++r) { mrun[r] = -__builtin_inff(); lrun[r] = 0.f; }

  const int vrow = tid >> 3, vchunk = tid & 7;  // V stage: 32 rows x 8 uint4

  for (int kb = 0; kb < L_SEQ; kb += 32) {
    // ---- stage V tile (32 keys x 64 d, bf16) coalesced into LDS
    {
      uint4 u = *(const uint4*)(V + (size_t)(kb + vrow) * HDIM + vchunk * 8);
      *(uint4*)&Vs[vrow][vchunk * 8] = u;
    }
    if (kb + 32 < L_SEQ)
      __builtin_prefetch(V + (size_t)(kb + 32 + vrow) * HDIM, 0, 0);

    // ---- S = Q K^T : 2 accumulators of 16 keys each (K read per-lane b128s)
    v8f s[2];
    s[0] = v8f_zero(); s[1] = v8f_zero();
#pragma unroll
    for (int j = 0; j < 2; ++j) {
#pragma unroll
      for (int ds = 0; ds < 2; ++ds) {
        v16bf bf;
        const __bf16* kp = K + (size_t)(kb + j * 16 + nl) * HDIM + ds * 32 + hi * 16;
#pragma unroll
        for (int e = 0; e < 16; ++e) bf[e] = kp[e];
        s[j] = wmma_bf16(qf[ds], bf, s[j]);
      }
    }

    // ---- online softmax (rows live as elem r + lane-half)
    float mnew[8], sc[8], pj0[8], pj1[8];
#pragma unroll
    for (int r = 0; r < 8; ++r) {
      float v = fmaxf(s[0][r], s[1][r]) * inv_scale;
      v = fmaxf(v, __shfl_xor(v, 1, 32));
      v = fmaxf(v, __shfl_xor(v, 2, 32));
      v = fmaxf(v, __shfl_xor(v, 4, 32));
      v = fmaxf(v, __shfl_xor(v, 8, 32));
      mnew[r] = fmaxf(mrun[r], v);
      sc[r] = __expf(mrun[r] - mnew[r]);
      float p0 = __expf(s[0][r] * inv_scale - mnew[r]);
      float p1 = __expf(s[1][r] * inv_scale - mnew[r]);
      float rs = p0 + p1;
      rs += __shfl_xor(rs, 1, 32);
      rs += __shfl_xor(rs, 2, 32);
      rs += __shfl_xor(rs, 4, 32);
      rs += __shfl_xor(rs, 8, 32);
      lrun[r] = lrun[r] * sc[r] + rs;
      mrun[r] = mnew[r];
      pj0[r] = p0; pj1[r] = p1;
    }
#pragma unroll
    for (int nd = 0; nd < 4; ++nd)
#pragma unroll
      for (int r = 0; r < 8; ++r) o[nd][r] *= sc[r];

    // ---- C-layout P -> LDS (A-relayout) ; barrier also publishes Vs
#pragma unroll
    for (int r = 0; r < 8; ++r) {
      plds[wave][r + 8 * hi][nl] = (__bf16)pj0[r];
      plds[wave][r + 8 * hi][16 + nl] = (__bf16)pj1[r];
    }
    __syncthreads();
    v16bf pf;
    {
      const __bf16* p = &plds[wave][nl][0];
#pragma unroll
      for (int e = 0; e < 8; ++e) pf[e] = p[hi * 16 + e];
#pragma unroll
      for (int e = 0; e < 8; ++e) pf[8 + e] = p[hi * 16 + 8 + e];
    }

    // ---- O += P V : V fragments gathered from LDS (stride = 72 elems)
#pragma unroll
    for (int nd = 0; nd < 4; ++nd) {
      v16bf vf;
#pragma unroll
      for (int e = 0; e < 16; ++e) vf[e] = Vs[hi * 16 + e][nd * 16 + nl];
      o[nd] = wmma_bf16(pf, vf, o[nd]);
    }
    __syncthreads();
  }

  // normalize + scatter back to model layout [b][l][h*64+d] bf16
  const int bb = bh >> 4, h = bh & 15;
  __bf16* ob = obuf + (size_t)branch * HBELEMS;
#pragma unroll
  for (int nd = 0; nd < 4; ++nd)
#pragma unroll
    for (int r = 0; r < 8; ++r) {
      int l = qrow0 + r + 8 * hi;
      int d = nd * 16 + nl;
      float val = o[nd][r] / lrun[r];
      ob[((size_t)bb * L_SEQ + l) * HALFD + h * HDIM + d] = (__bf16)val;
    }
}

// ================= Kernel 3: output projection (bf16 A) =========================
// P = O(4096x1024 bf16) @ Wo^T -> f32. Block tile 128x128. A tile optionally
// DMA'd by the Tensor Data Mover (pad fields reproduce the LDS_STRIDE=40 layout).
__global__ __launch_bounds__(256) void out_proj_kernel(
    const __bf16* __restrict__ obuf, const float* __restrict__ Woa,
    const float* __restrict__ Wob, float* __restrict__ pbuf) {
  __shared__ __align__(16) __bf16 As[BM][LDS_STRIDE];
  __shared__ __align__(16) __bf16 Bs[BN][LDS_STRIDE];

  const int branch = blockIdx.z;
  const __bf16* X = obuf + (size_t)branch * HBELEMS;
  const float* W = branch ? Wob : Woa;
  float* out = pbuf + (size_t)branch * HBELEMS;

  const int m0 = blockIdx.y * BM;
  const int n0 = blockIdx.x * BN;
  const int tid = threadIdx.x, lane = tid & 31, wave = tid >> 5;
  const int wm = wave >> 1, wn = wave & 1;

  v8f acc[2][4];
#pragma unroll
  for (int i = 0; i < 2; ++i)
#pragma unroll
    for (int j = 0; j < 4; ++j) acc[i][j] = v8f_zero();

#if HAVE_TDM
  // TDM descriptor group 1 (constant across K loop): 2D tensor, 2-byte elems,
  // tile 32(k) x 128(rows), LDS padding 4 DWORDs every 16 DWORDs -> stride 40.
  v8i_ g1;
  g1[0] = (1 << 16) | (1 << 20) | (3 << 22) | (3 << 25);
  g1[1] = (int)((HALFD & 0xFFFF) << 16);           // tensor_dim0 lo
  g1[2] = (int)((HALFD >> 16) | ((MTOT & 0xFFFF) << 16));
  g1[3] = (int)((MTOT >> 16) | (BK << 16));        // tile_dim0 = 32
  g1[4] = BM;                                      // tile_dim1 = 128
  g1[5] = HALFD;                                   // tensor_dim0_stride
  g1[6] = 0; g1[7] = 0;
  v4i_ zz = {0, 0, 0, 0};
  const unsigned lds_as = (unsigned)(size_t)&As[0][0];
#endif

  for (int k0 = 0; k0 < HALFD; k0 += BK) {
#if HAVE_TDM
    if (wave == 0) {
      unsigned long long ga =
          (unsigned long long)(size_t)(X + (size_t)m0 * HALFD + k0);
      v4u g0;
      g0.x = 1u;  // count = 1 valid descriptor
      g0.y = lds_as;
      g0.z = (unsigned)ga;
      g0.w = (unsigned)((ga >> 32) & 0x01FFFFFFu) | (2u << 30);  // type=2
#if __clang_major__ >= 23
      v8i_ z8 = {0, 0, 0, 0, 0, 0, 0, 0};
      __builtin_amdgcn_tensor_load_to_lds(g0, g1, zz, zz, z8, 0);
#else
      __builtin_amdgcn_tensor_load_to_lds(g0, g1, zz, zz, 0);
#endif
    }
#else
    {  // cooperative A tile stage: 128x32 bf16 via uint4 (8 bf16 each)
      int tc = tid & 3, tr = tid >> 2;  // 4 chunks/row, 64 rows/pass
#pragma unroll
      for (int p = 0; p < 2; ++p) {
        int r = tr + p * 64;
        uint4 u = *(const uint4*)(X + (size_t)(m0 + r) * HALFD + k0 + tc * 8);
        *(uint4*)&As[r][tc * 8] = u;
      }
    }
#endif
    {  // B tile from Wo (f32 -> bf16), 128 n-rows x 32 k
      int tc = tid & 7, tr = tid >> 3;
#pragma unroll
      for (int p = 0; p < 4; ++p) {
        int r = tr + p * 32;
        float4 v = *(const float4*)(W + (size_t)(n0 + r) * HALFD + k0 + tc * 4);
        __bf16* dst = &Bs[r][tc * 4];
        dst[0] = (__bf16)v.x; dst[1] = (__bf16)v.y;
        dst[2] = (__bf16)v.z; dst[3] = (__bf16)v.w;
      }
    }
#if HAVE_TDM
    if (wave == 0) __builtin_amdgcn_s_wait_tensorcnt(0);
#endif
    __syncthreads();

    v16bf bfr[4];
#pragma unroll
    for (int ni = 0; ni < 4; ++ni)
      bfr[ni] = load_b_frag(&Bs[wn * 64 + ni * 16][0], LDS_STRIDE);
#pragma unroll
    for (int mi = 0; mi < 2; ++mi) {
      v16bf a = load_a_frag(&As[wm * 32 + mi * 16][0], LDS_STRIDE);
#pragma unroll
      for (int ni = 0; ni < 4; ++ni)
        acc[mi][ni] = wmma_bf16(a, bfr[ni], acc[mi][ni]);
    }
    __syncthreads();
  }

  const int nl = lane & 15, hi = lane >> 4;
#pragma unroll
  for (int mi = 0; mi < 2; ++mi)
#pragma unroll
    for (int ni = 0; ni < 4; ++ni)
#pragma unroll
      for (int r = 0; r < 8; ++r) {
        int gm = m0 + wm * 32 + mi * 16 + r + 8 * hi;
        int gn = n0 + wn * 64 + ni * 16 + nl;
        out[(size_t)gm * HALFD + gn] = acc[mi][ni][r];
      }
}

// ================= Kernel 4: residual + LayerNorm ===============================
__global__ __launch_bounds__(256) void ln_kernel(
    const float* __restrict__ pbuf, const float* __restrict__ xa,
    const float* __restrict__ xb, const float* __restrict__ ga,
    const float* __restrict__ ba, const float* __restrict__ gb,
    const float* __restrict__ bb, float* __restrict__ out) {
  const int row = blockIdx.x;
  const int branch = blockIdx.y;
  const float* P = pbuf + (size_t)branch * HBELEMS;
  const float* X = branch ? xb : xa;
  const float* G = branch ? gb : ga;
  const float* Bt = branch ? bb : ba;

  __shared__ float red[2][8];
  float v[4], s1 = 0.f, s2 = 0.f;
#pragma unroll
  for (int i = 0; i < 4; ++i) {
    int c = threadIdx.x + i * 256;
    float t = P[(size_t)row * HALFD + c] + X[(size_t)row * HALFD + c];
    v[i] = t; s1 += t; s2 += t * t;
  }
#pragma unroll
  for (int off = 1; off < 32; off <<= 1) {
    s1 += __shfl_xor(s1, off, 32);
    s2 += __shfl_xor(s2, off, 32);
  }
  int lane = threadIdx.x & 31, wave = threadIdx.x >> 5;
  if (lane == 0) { red[0][wave] = s1; red[1][wave] = s2; }
  __syncthreads();
  float S1 = 0.f, S2 = 0.f;
#pragma unroll
  for (int w = 0; w < 8; ++w) { S1 += red[0][w]; S2 += red[1][w]; }
  float mu = S1 * (1.f / 1024.f);
  float var = S2 * (1.f / 1024.f) - mu * mu;
  float rstd = rsqrtf(var + 1e-5f);
#pragma unroll
  for (int i = 0; i < 4; ++i) {
    int c = threadIdx.x + i * 256;
    out[(size_t)branch * HBELEMS + (size_t)row * HALFD + c] =
        (v[i] - mu) * rstd * G[c] + Bt[c];
  }
}

// ================================ Host launch ===================================
extern "C" void kernel_launch(void* const* d_in, const int* in_sizes, int n_in,
                              void* d_out, int out_size, void* d_ws, size_t ws_size,
                              hipStream_t stream) {
  (void)in_sizes; (void)n_in; (void)out_size; (void)ws_size;
  const float* x_a = (const float*)d_in[0];
  const float* x_b = (const float*)d_in[1];
  const float* Wq_a = (const float*)d_in[2];
  const float* Wq_b = (const float*)d_in[3];
  const float* Wk_a = (const float*)d_in[4];
  const float* Wk_b = (const float*)d_in[5];
  const float* Wv_a = (const float*)d_in[6];
  const float* Wv_b = (const float*)d_in[7];
  const float* Wo_a = (const float*)d_in[8];
  const float* Wo_b = (const float*)d_in[9];
  const float* g_a = (const float*)d_in[10];
  const float* b_a = (const float*)d_in[11];
  const float* g_b = (const float*)d_in[12];
  const float* b_b = (const float*)d_in[13];

  // workspace (~96 MiB, fully L2-resident on MI455X):
  // [0,48M): 6 bf16 QKV buffers head-major | [48M,64M): 2 bf16 attn outputs
  // [64M,96M): 2 f32 projection buffers
  __bf16* qkv = (__bf16*)d_ws;
  __bf16* obuf = qkv + 6ull * HBELEMS;
  float* pbuf = (float*)(obuf + 2ull * HBELEMS);
  float* out = (float*)d_out;

  qkv_proj_kernel<<<dim3(HALFD / BN, MTOT / BM, 6), 256, 0, stream>>>(
      x_a, x_b, Wq_a, Wk_a, Wv_a, Wq_b, Wk_b, Wv_b, qkv);
  cross_attn_kernel<<<dim3(L_SEQ / 128, 32, 2), 256, 0, stream>>>(qkv, obuf);
  out_proj_kernel<<<dim3(HALFD / BN, MTOT / BM, 2), 256, 0, stream>>>(
      obuf, Wo_a, Wo_b, pbuf);
  ln_kernel<<<dim3(MTOT, 2), 256, 0, stream>>>(pbuf, x_a, x_b, g_a, b_a, g_b, b_b,
                                               out);
}
